// KBGATModel_8933531976193
// MI455X (gfx1250) — matrix operand order repaired and checked by reading, not compiled, not run on hardware
//
#include <hip/hip_runtime.h>
#include <hip/hip_bf16.h>
#include <float.h>
#include <math.h>

// ---------------- problem constants ----------------
#define NENT   40000
#define NRELS  500
#define NEDGE  400000
#define EDIM   50
#define RDIM   50
#define NHEADS 8
#define U0     150
#define U1     300
#define NEG_SLOPE 0.2f

// padded GEMM shapes (K mult of 32, N mult of 32; M = 40000 = 1250*32)
#define K0P    64      // EDIM 50 -> 64
#define N0P    320     // 2*U0 300 -> 320
#define K1P    1216    // 8*U0 1200 -> 1216
#define N1P    608     // 2*U1 600 -> 608 (19*32)

typedef __attribute__((ext_vector_type(16))) __bf16 v16bf;
typedef __attribute__((ext_vector_type(8)))  float  v8f;

// ---------------- helpers ----------------
__device__ __forceinline__ unsigned f2ord(float f) {
    unsigned u = __float_as_uint(f);
    return (u & 0x80000000u) ? ~u : (u | 0x80000000u);
}
__device__ __forceinline__ float ord2f(unsigned u) {
    return (u & 0x80000000u) ? __uint_as_float(u & 0x7FFFFFFFu) : __uint_as_float(~u);
}
__device__ __forceinline__ float eluf(float x) { return x > 0.f ? x : (expf(x) - 1.f); }

// Swizzled fragment layouts: flat index i = ((tile*kblocks + kb)*32 + lane)*16 + j
// A-frag:  k = kb*32 + (lane>>4)*8 + (j<8 ? j : j+8), row = tile*16 + (lane&15)
// B-frag:  k = kb*32 + (lane>>4)*8 + (j<8 ? j : j+8),              col = tile*16 + (lane&15)

// ---------------- utility kernels ----------------
__global__ void k_fill_u32(unsigned* __restrict__ p, unsigned v, int n) {
    int i = blockIdx.x * blockDim.x + threadIdx.x;
    if (i < n) p[i] = v;
}

// entity embeddings -> swizzled A fragments [NENT x K0P], kblocks = 2
__global__ void k_pack_entA(const float* __restrict__ e, __bf16* __restrict__ o, int n) {
    int i = blockIdx.x * blockDim.x + threadIdx.x;
    if (i >= n) return;
    int j = i & 15, lane = (i >> 4) & 31, blk = i >> 9;
    int kb = blk & 1, mt = blk >> 1;                       // kblocks = 2
    int k   = kb * 32 + (lane >> 4) * 8 + (j < 8 ? j : j + 8);
    int row = mt * 16 + (lane & 15);
    o[i] = (__bf16)(k < EDIM ? e[(size_t)row * EDIM + k] : 0.f);
}

// gat0_kernel[hd] rows: Wh=0..49, Wt=100..149 -> swizzled B frags [K0P x N0P]
__global__ void k_pack_B0(const float* __restrict__ W, __bf16* __restrict__ B, int n) {
    int i = blockIdx.x * blockDim.x + threadIdx.x;
    if (i >= n) return;
    const int kblocks = K0P / 32;
    int j = i & 15, lane = (i >> 4) & 31, blk = i >> 9;
    int kb = blk % kblocks, nt = blk / kblocks;
    int k = kb * 32 + (lane >> 4) * 8 + (j < 8 ? j : j + 8);
    int c = nt * 16 + (lane & 15);
    float v = 0.f;
    if (k < EDIM) {
        if (c < U0)          v = W[k * 150 + c];
        else if (c < 2 * U0) v = W[(100 + k) * 150 + (c - U0)];
    }
    B[i] = (__bf16)v;
}

// gat1_kernel[hd] rows: Wh=0..1199, Wt=1350..2549 -> swizzled B frags [K1P x N1P]
__global__ void k_pack_B1(const float* __restrict__ W, __bf16* __restrict__ B, int n) {
    int i = blockIdx.x * blockDim.x + threadIdx.x;
    if (i >= n) return;
    const int kblocks = K1P / 32;
    int j = i & 15, lane = (i >> 4) & 31, blk = i >> 9;
    int kb = blk % kblocks, nt = blk / kblocks;
    int k = kb * 32 + (lane >> 4) * 8 + (j < 8 ? j : j + 8);
    int c = nt * 16 + (lane & 15);
    float v = 0.f;
    if (k < 1200) {
        if (c < U1)          v = W[(size_t)k * U1 + c];
        else if (c < 2 * U1) v = W[(size_t)(1350 + k) * U1 + (c - U1)];
    }
    B[i] = (__bf16)v;
}

// ---------------- WMMA GEMM on swizzled fragments ----------------
// wave computes a 32(M) x 32(N) tile: 2 A-frags x 2 B-frags -> 4 accumulators.
// grid = (N/32, M/32), block = 32 threads.
__global__ void k_gemm_bf16(const __bf16* __restrict__ Ap, const __bf16* __restrict__ Bp,
                            float* __restrict__ C, int N, int K) {
    const int lane = threadIdx.x;            // 0..31
    const int kblocks = K >> 5;
    const int mt0 = blockIdx.y * 2;
    const int nt0 = blockIdx.x * 2;

    const v16bf* a0 = (const v16bf*)Ap + (size_t)mt0 * kblocks * 32 + lane;
    const v16bf* a1 = a0 + (size_t)kblocks * 32;
    const v16bf* b0 = (const v16bf*)Bp + (size_t)nt0 * kblocks * 32 + lane;
    const v16bf* b1 = b0 + (size_t)kblocks * 32;

    v8f acc00 = {}, acc01 = {}, acc10 = {}, acc11 = {};
    for (int kb = 0; kb < kblocks; ++kb) {
        const size_t o = (size_t)kb * 32;
        __builtin_prefetch(a0 + o + 32, 0, 3);    // global_prefetch_b8
        __builtin_prefetch(a1 + o + 32, 0, 3);
        v16bf fa0 = a0[o];
        v16bf fa1 = a1[o];
        v16bf fb0 = b0[o];
        v16bf fb1 = b1[o];
        acc00 = __builtin_amdgcn_wmma_f32_16x16x32_bf16(false, fa0, false, fb0,
                                                        (short)0, acc00, false, false);
        acc01 = __builtin_amdgcn_wmma_f32_16x16x32_bf16(false, fa0, false, fb1,
                                                        (short)0, acc01, false, false);
        acc10 = __builtin_amdgcn_wmma_f32_16x16x32_bf16(false, fa1, false, fb0,
                                                        (short)0, acc10, false, false);
        acc11 = __builtin_amdgcn_wmma_f32_16x16x32_bf16(false, fa1, false, fb1,
                                                        (short)0, acc11, false, false);
    }
    // D layout: VGPR v, lanes 0-15 -> M=v, lanes 16-31 -> M=v+8; N=lane&15
    const int ml = lane & 15, half = lane >> 4;
    float* c00 = C + ((size_t)mt0 * 16 + half * 8) * N + nt0 * 16 + ml;
    float* c10 = c00 + (size_t)16 * N;
#pragma unroll
    for (int v = 0; v < 8; ++v) {
        c00[(size_t)v * N]      = acc00[v];
        c00[(size_t)v * N + 16] = acc01[v];
        c10[(size_t)v * N]      = acc10[v];
        c10[(size_t)v * N + 16] = acc11[v];
    }
}

// ---------------- small dense f32 matmul (C = A@B (+bias)) ----------------
__global__ void k_matmul_f32(const float* __restrict__ A, const float* __restrict__ B,
                             const float* __restrict__ bias, float* __restrict__ C,
                             int M, int N, int K, int ldb) {
    int i = blockIdx.x * blockDim.x + threadIdx.x;
    if (i >= M * N) return;
    int m = i / N, n = i % N;
    float acc = bias ? bias[n] : 0.f;
    const float* a = A + (size_t)m * K;
    for (int k = 0; k < K; ++k) acc += a[k] * B[(size_t)k * ldb + n];
    C[i] = acc;
}

// ---------------- edge kernels (wave per edge) ----------------
__global__ void k_edge_score(const float* __restrict__ PHPT, const float* __restrict__ PR,
                             const float* __restrict__ attn,
                             const int* __restrict__ h, const int* __restrict__ r,
                             const int* __restrict__ t,
                             float* __restrict__ sbuf, unsigned* __restrict__ segmax,
                             int E, int U, int strideP, int toff) {
    int wid  = (blockIdx.x * blockDim.x + threadIdx.x) >> 5;
    int lane = threadIdx.x & 31;
    if (wid >= E) return;
    int hi = h[wid], ri = r[wid], ti = t[wid];
    const float* ph = PHPT + (size_t)hi * strideP;
    const float* pt = PHPT + (size_t)ti * strideP + toff;
    const float* pr = PR   + (size_t)ri * U;
    float acc = 0.f;
    for (int u = lane; u < U; u += 32)
        acc += (ph[u] + pr[u] + pt[u]) * attn[u];
    for (int off = 16; off > 0; off >>= 1) acc += __shfl_down(acc, off, 32);
    if (lane == 0) {
        float s = acc > 0.f ? acc : NEG_SLOPE * acc;
        sbuf[wid] = s;
        atomicMax(segmax + hi, f2ord(s));
    }
}

__global__ void k_edge_exp(const float* __restrict__ sbuf, const unsigned* __restrict__ segmax,
                           const int* __restrict__ h, float* __restrict__ ebuf,
                           float* __restrict__ denom, int E) {
    int e = blockIdx.x * blockDim.x + threadIdx.x;
    if (e >= E) return;
    int hi = h[e];
    float ex = expf(sbuf[e] - ord2f(segmax[hi]));
    ebuf[e] = ex;
    atomicAdd(denom + hi, ex);
}

__global__ void k_edge_agg(const float* __restrict__ PHPT, const float* __restrict__ PR,
                           const float* __restrict__ ebuf, const float* __restrict__ denom,
                           const int* __restrict__ h, const int* __restrict__ r,
                           const int* __restrict__ t, float* __restrict__ agg,
                           int E, int U, int strideP, int toff) {
    int wid  = (blockIdx.x * blockDim.x + threadIdx.x) >> 5;
    int lane = threadIdx.x & 31;
    if (wid >= E) return;
    int hi = h[wid], ri = r[wid], ti = t[wid];
    float a = ebuf[wid] / (denom[hi] + 1e-10f);
    const float* ph = PHPT + (size_t)hi * strideP;
    const float* pt = PHPT + (size_t)ti * strideP + toff;
    const float* pr = PR   + (size_t)ri * U;
    float* dst = agg + (size_t)hi * U;
    for (int u = lane; u < U; u += 32)
        atomicAdd(dst + u, a * (ph[u] + pr[u] + pt[u]));
}

// elu(agg0) -> swizzled feat A-fragment buffer (kblocks = K1P/32), column block hd*U0
__global__ void k_elu_feat(const float* __restrict__ agg, __bf16* __restrict__ feat,
                           int n, int coff) {
    int i = blockIdx.x * blockDim.x + threadIdx.x;
    if (i >= n) return;
    const int kblocks = K1P / 32;
    int row = i / U0, u = i % U0;
    int k = coff + u;
    int mt = row >> 4, ml = row & 15;
    int kb = k >> 5, rem = k & 31;
    int half = (rem >> 3) & 1;
    int j = (rem & 7) + (rem >= 16 ? 8 : 0);
    size_t idx = ((size_t)(mt * kblocks + kb) * 32 + half * 16 + ml) * 16 + j;
    feat[idx] = (__bf16)eluf(agg[i]);
}

// ent_out += elu(agg1)
__global__ void k_add_elu(const float* __restrict__ agg, float* __restrict__ out, int n) {
    int i = blockIdx.x * blockDim.x + threadIdx.x;
    if (i < n) out[i] += eluf(agg[i]);
}

// ---------------- host orchestration ----------------
static inline int ceildiv(long long a, long long b) { return (int)((a + b - 1) / b); }

extern "C" void kernel_launch(void* const* d_in, const int* in_sizes, int n_in,
                              void* d_out, int out_size, void* d_ws, size_t ws_size,
                              hipStream_t stream) {
    (void)in_sizes; (void)n_in; (void)out_size; (void)ws_size;
    const int*   h_idx = (const int*)d_in[0];
    const int*   r_idx = (const int*)d_in[1];
    const int*   t_idx = (const int*)d_in[2];
    const float* ent   = (const float*)d_in[3];
    const float* rel   = (const float*)d_in[4];
    const float* W0    = (const float*)d_in[5];   // [8,150,150]
    const float* attn0 = (const float*)d_in[6];   // [8,150]
    const float* relk0 = (const float*)d_in[7];   // [50,150]
    const float* W1    = (const float*)d_in[8];   // [8,2550,300]
    const float* attn1 = (const float*)d_in[9];   // [8,300]
    const float* relk1 = (const float*)d_in[10];  // [150,300]
    const float* dw    = (const float*)d_in[11];  // [50,300]
    const float* db    = (const float*)d_in[12];  // [300]
    float* out = (float*)d_out;                   // [40000*300] ++ [500*300]

    // bump allocator on workspace
    char*  base = (char*)d_ws;
    size_t off  = 0;
    auto alloc = [&](size_t bytes) -> void* {
        off = (off + 255) & ~(size_t)255;
        void* p = base + off;
        off += bytes;
        return p;
    };
    __bf16*   entb   = (__bf16*)  alloc((size_t)NENT * K0P * 2);
    __bf16*   featb  = (__bf16*)  alloc((size_t)NENT * K1P * 2);
    __bf16*   Bb     = (__bf16*)  alloc((size_t)K1P * N1P * 2);
    float*    PHPT   = (float*)   alloc((size_t)NENT * N1P * 4);
    float*    PR     = (float*)   alloc((size_t)NRELS * U1 * 4);
    float*    rel0   = (float*)   alloc((size_t)NRELS * U0 * 4);
    float*    sbuf   = (float*)   alloc((size_t)NEDGE * 4);
    float*    ebuf   = (float*)   alloc((size_t)NEDGE * 4);
    unsigned* segmax = (unsigned*)alloc((size_t)NENT * 4);
    float*    denom  = (float*)   alloc((size_t)NENT * 4);
    float*    agg    = (float*)   alloc((size_t)NENT * U1 * 4);

    const int B256 = 256;
    const unsigned ORD_NEG_MAX = 0x00800000u;   // f2ord(-FLT_MAX)

    // ---- packs / init ----
    k_pack_entA<<<ceildiv((long long)NENT * K0P, B256), B256, 0, stream>>>(ent, entb, NENT * K0P);
    k_fill_u32<<<ceildiv((long long)NENT * K1P / 2, B256), B256, 0, stream>>>(
        (unsigned*)featb, 0u, NENT * K1P / 2);   // zero feat (incl. padding)

    const int edgeWaveBlocks = ceildiv((long long)NEDGE * 32, B256);

    // ================= layer 0, per head =================
    for (int hd = 0; hd < NHEADS; ++hd) {
        const float* W0h = W0 + (size_t)hd * 150 * 150;
        k_pack_B0<<<ceildiv(K0P * N0P, B256), B256, 0, stream>>>(W0h, Bb, K0P * N0P);
        dim3 g0(N0P / 32, NENT / 32);
        k_gemm_bf16<<<g0, 32, 0, stream>>>(entb, Bb, PHPT, N0P, K0P);
        // PR0 = rel @ Wr  (Wr = rows 50..99 of W0h)
        k_matmul_f32<<<ceildiv(NRELS * U0, B256), B256, 0, stream>>>(
            rel, W0h + 50 * 150, nullptr, PR, NRELS, U0, RDIM, 150);
        k_fill_u32<<<ceildiv(NENT, B256), B256, 0, stream>>>(segmax, ORD_NEG_MAX, NENT);
        k_fill_u32<<<ceildiv(NENT, B256), B256, 0, stream>>>((unsigned*)denom, 0u, NENT);
        k_fill_u32<<<ceildiv(NENT * U0, B256), B256, 0, stream>>>((unsigned*)agg, 0u, NENT * U0);
        k_edge_score<<<edgeWaveBlocks, B256, 0, stream>>>(
            PHPT, PR, attn0 + hd * U0, h_idx, r_idx, t_idx, sbuf, segmax, NEDGE, U0, N0P, U0);
        k_edge_exp<<<ceildiv(NEDGE, B256), B256, 0, stream>>>(sbuf, segmax, h_idx, ebuf, denom, NEDGE);
        k_edge_agg<<<edgeWaveBlocks, B256, 0, stream>>>(
            PHPT, PR, ebuf, denom, h_idx, r_idx, t_idx, agg, NEDGE, U0, N0P, U0);
        k_elu_feat<<<ceildiv(NENT * U0, B256), B256, 0, stream>>>(agg, featb, NENT * U0, hd * U0);
    }

    // rel0 = rel @ gat0_rel_kernel
    k_matmul_f32<<<ceildiv(NRELS * U0, B256), B256, 0, stream>>>(
        rel, relk0, nullptr, rel0, NRELS, U0, RDIM, U0);
    // ent_out = ent @ dense_w + dense_b
    k_matmul_f32<<<ceildiv(NENT * U1, B256), B256, 0, stream>>>(
        ent, dw, db, out, NENT, U1, EDIM, U1);

    // ================= layer 1, per head =================
    for (int hd = 0; hd < NHEADS; ++hd) {
        const float* W1h = W1 + (size_t)hd * 2550 * 300;
        k_pack_B1<<<ceildiv(K1P * N1P, B256), B256, 0, stream>>>(W1h, Bb, K1P * N1P);
        dim3 g1(N1P / 32, NENT / 32);
        k_gemm_bf16<<<g1, 32, 0, stream>>>(featb, Bb, PHPT, N1P, K1P);
        // PR1 = rel0 @ Wr1  (rows 1200..1349 of W1h)
        k_matmul_f32<<<ceildiv(NRELS * U1, B256), B256, 0, stream>>>(
            rel0, W1h + (size_t)1200 * 300, nullptr, PR, NRELS, U1, U0, 300);
        k_fill_u32<<<ceildiv(NENT, B256), B256, 0, stream>>>(segmax, ORD_NEG_MAX, NENT);
        k_fill_u32<<<ceildiv(NENT, B256), B256, 0, stream>>>((unsigned*)denom, 0u, NENT);
        k_fill_u32<<<ceildiv(NENT * U1, B256), B256, 0, stream>>>((unsigned*)agg, 0u, NENT * U1);
        k_edge_score<<<edgeWaveBlocks, B256, 0, stream>>>(
            PHPT, PR, attn1 + hd * U1, h_idx, r_idx, t_idx, sbuf, segmax, NEDGE, U1, N1P, U1);
        k_edge_exp<<<ceildiv(NEDGE, B256), B256, 0, stream>>>(sbuf, segmax, h_idx, ebuf, denom, NEDGE);
        k_edge_agg<<<edgeWaveBlocks, B256, 0, stream>>>(
            PHPT, PR, ebuf, denom, h_idx, r_idx, t_idx, agg, NEDGE, U1, N1P, U1);
        k_add_elu<<<ceildiv(NENT * U1, B256), B256, 0, stream>>>(agg, out, NENT * U1);
    }

    // rel1 = rel0 @ gat1_rel_kernel -> tail of output
    k_matmul_f32<<<ceildiv(NRELS * U1, B256), B256, 0, stream>>>(
        rel0, relk1, nullptr, out + (size_t)NENT * U1, NRELS, U1, U0, U1);
}
